// VarFlowLoss_61057255080539
// MI455X (gfx1250) — matrix operand order repaired and checked by reading, not compiled, not run on hardware
//
#include <hip/hip_runtime.h>
#include <hip/hip_bf16.h>

typedef __attribute__((ext_vector_type(16))) __bf16          v16bf;
typedef __attribute__((ext_vector_type(8)))  float           v8f;
typedef __attribute__((ext_vector_type(8)))  unsigned short  ushort8;
typedef __attribute__((ext_vector_type(16))) unsigned short  ushort16;

#define ENT   198     // 3*66 halo entries per row-strip
#define ENT_B 80      // bytes per LDS entry: 64B ic-chunk + 16B pad (16B-aligned, conflict-free b128 reads)
#define ENT_H 40      // halfs per LDS entry

__device__ __forceinline__ unsigned short f2bf(float f) {
    unsigned u = __float_as_uint(f);
    u += 0x7FFFu + ((u >> 16) & 1u);          // round-to-nearest-even
    return (unsigned short)(u >> 16);
}

__device__ __forceinline__ v16bf make_frag(const unsigned short* p0, const unsigned short* p1) {
    ushort8 lo = *(const ushort8*)p0;
    ushort8 hi = *(const ushort8*)p1;
    ushort16 full = __builtin_shufflevector(lo, hi, 0,1,2,3,4,5,6,7,8,9,10,11,12,13,14,15);
    return __builtin_bit_cast(v16bf, full);
}

// CDNA5 async global->LDS copy, 16B per lane, tracked by ASYNCcnt.
__device__ __forceinline__ void async_cp16(unsigned lds_addr, const void* gaddr) {
    asm volatile("global_load_async_to_lds_b128 %0, %1, off"
                 :: "v"(lds_addr), "v"(gaddr) : "memory");
}
__device__ __forceinline__ void wait_async0() {
    asm volatile("s_wait_asynccnt 0x0" ::: "memory");
}

// ---------------- feat NCHW f32 -> NHWC bf16 (featT[b][y][x][ic]) ----------------
__global__ __launch_bounds__(256) void k_nhwc(const float* __restrict__ feat,
                                              unsigned short* __restrict__ featT) {
    __shared__ __align__(16) unsigned short t[64 * 264];   // t[x*264 + ic], padded
    const int y = blockIdx.x, b = blockIdx.y;
    const float* src = feat + (long)b * 256 * 4096 + y * 64;
    for (int idx = threadIdx.x; idx < 256 * 64; idx += 256) {
        int ic = idx >> 6, x = idx & 63;                   // consecutive tid -> consecutive x (coalesced)
        t[x * 264 + ic] = f2bf(src[(long)ic * 4096 + x]);
    }
    __syncthreads();
    unsigned short* dst = featT + (((long)b * 64 + y) * 64) * 256;
    for (int p = threadIdx.x; p < 64 * 32; p += 256) {     // 16B vector stores, ic contiguous
        int x = p >> 5, g = p & 31;
        *(ushort8*)&dst[x * 256 + g * 8] = *(const ushort8*)&t[x * 264 + g * 8];
    }
}

// ---------------- W_att[oc][ic][3][3] f32 -> WattT[tap][oc][ic] bf16 ----------------
__global__ void k_prep_watt(const float* __restrict__ W, unsigned short* __restrict__ Wt) {
    int i = blockIdx.x * 256 + threadIdx.x;
    if (i >= 9 * 64 * 256) return;
    int tap = i / (64 * 256);
    int rem = i - tap * 64 * 256;
    int oc  = rem >> 8;
    int ic  = rem & 255;
    Wt[i] = f2bf(W[(oc * 256 + ic) * 9 + tap]);
}

// ---------------- W_post[10][64][3][3] f32 -> WpostT[tap][n(16,zero-pad)][ic] bf16 ----------------
__global__ void k_prep_wpost(const float* __restrict__ W, unsigned short* __restrict__ Wt) {
    int i = blockIdx.x * 256 + threadIdx.x;
    if (i >= 9 * 16 * 64) return;
    int tap = i / (16 * 64);
    int rem = i - tap * 1024;
    int n   = rem >> 6;
    int ic  = rem & 63;
    Wt[i] = (n < 10) ? f2bf(W[(n * 64 + ic) * 9 + tap]) : (unsigned short)0;
}

// ---------------- per-patch border log-sums of gts[32,1,512,512] ----------------
__global__ void k_border(const float* __restrict__ gts,
                         float* __restrict__ Bn, float* __restrict__ Bs,
                         float* __restrict__ Bw, float* __restrict__ Be) {
    int i = blockIdx.x * 256 + threadIdx.x;
    if (i >= 32 * 64 * 64) return;
    int b  = i >> 12;
    int py = (i >> 6) & 63;
    int px = i & 63;
    const float* base = gts + ((long)b * 512 + py * 8) * 512 + px * 8;
    float sn = 0.f, ss = 0.f, sw = 0.f, se = 0.f;
#pragma unroll
    for (int t = 0; t < 8; ++t) {
        float vn = base[t];
        float vs = base[7 * 512 + t];
        float vw = base[t * 512];
        float ve = base[t * 512 + 7];
        sn += __logf((vn > 0.1f && vn < 80.f ? vn : 0.f) + 1e-6f);
        ss += __logf((vs > 0.1f && vs < 80.f ? vs : 0.f) + 1e-6f);
        sw += __logf((vw > 0.1f && vw < 80.f ? vw : 0.f) + 1e-6f);
        se += __logf((ve > 0.1f && ve < 80.f ? ve : 0.f) + 1e-6f);
    }
    Bn[i] = sn; Bs[i] = ss; Bw[i] = sw; Be[i] = se;
}

// ---------------- flow[b][10][py][px] from border sums ----------------
__global__ void k_flow(const float* __restrict__ Bn, const float* __restrict__ Bs,
                       const float* __restrict__ Bw, const float* __restrict__ Be,
                       float* __restrict__ flow) {
    int i = blockIdx.x * 256 + threadIdx.x;
    if (i >= 32 * 64 * 64) return;
    int b  = i >> 12;
    int py = (i >> 6) & 63;
    int px = i & 63;
    float n0 = Bn[i], s0 = Bs[i], w0 = Bw[i], e0 = Be[i];
    bool hx = (px < 63), hy = (py < 63);
    float nx = hx ? Bn[i + 1]  : 0.f,  sx = hx ? Bs[i + 1]  : 0.f;
    float wx = hx ? Bw[i + 1]  : 0.f,  ex = hx ? Be[i + 1]  : 0.f;
    float ny = hy ? Bn[i + 64] : 0.f,  sy = hy ? Bs[i + 64] : 0.f;
    float wy = hy ? Bw[i + 64] : 0.f,  ey = hy ? Be[i + 64] : 0.f;
    float d[10] = { e0 - wx, s0 - ny, e0 - ex, s0 - sy, w0 - wx,
                    n0 - ny, s0 - sx, e0 - ey, n0 - nx, w0 - wy };
    float* out = flow + (long)b * 10 * 4096 + py * 64 + px;
#pragma unroll
    for (int j = 0; j < 10; ++j) out[j * 4096] = d[j] * 0.125f;
}

// ---------------- conv1: async-staged double-buffered implicit GEMM (bf16 WMMA) ----------------
__global__ __launch_bounds__(256) void k_conv1(const unsigned short* __restrict__ featT,
                                               const float* __restrict__ depth,
                                               const float* __restrict__ b_att,
                                               const unsigned short* __restrict__ WattT,
                                               const unsigned short* __restrict__ zerobuf,
                                               unsigned short* __restrict__ adT) {
    __shared__ __align__(16) unsigned char lbuf[2 * ENT * ENT_B + 16];   // ping-pong + dump slot
    const int tid = threadIdx.x, y = blockIdx.x, b = blockIdx.y;
    const int wv = tid >> 5, lane = tid & 31;
    const int m0 = (wv & 3) * 16, nb = (wv >> 2) * 32;
    const int ml = lane & 15, hh = lane >> 4;
    const unsigned lds_base = (unsigned)(unsigned long long)(void*)lbuf;
    const unsigned dump     = lds_base + 2u * ENT * ENT_B;
    const unsigned short* ls = (const unsigned short*)lbuf;

    // per-thread staging slots: 198 entries x 4 16B-parts, padded to 1024 lane-tasks
    unsigned sl[4];
    long     sg[4];
#pragma unroll
    for (int s = 0; s < 4; ++s) {
        int p = (wv * 4 + s) * 32 + lane;
        int e = p >> 2, q = p & 3;
        bool valid = (e < ENT);
        int r  = valid ? (e / 66) : 0;
        int cc = valid ? (e - r * 66) : 0;
        int yy = y - 1 + r, xx = cc - 1;
        bool inb = valid && ((unsigned)yy < 64u) && ((unsigned)xx < 64u);
        sl[s] = valid ? (lds_base + (unsigned)(e * ENT_B + q * 16)) : dump;
        sg[s] = inb ? ((((long)b * 64 + yy) * 64 + xx) * 256 + q * 8) : -1;
    }
    auto stage = [&](int bsel, int ic0) {
        unsigned boff = (unsigned)(bsel * ENT * ENT_B);
#pragma unroll
        for (int s = 0; s < 4; ++s) {
            const void* g = (sg[s] >= 0) ? (const void*)(featT + sg[s] + ic0)
                                         : (const void*)zerobuf;
            unsigned l = (sl[s] != dump) ? (sl[s] + boff) : dump;
            async_cp16(l, g);
        }
    };

    v8f c0 = {}, c1 = {};
    stage(0, 0);
    for (int k = 0; k < 8; ++k) {
        wait_async0();              // own copies for chunk k done
        __syncthreads();            // everyone's chunk-k data visible; prev compute finished
        if (k < 7) stage((k + 1) & 1, (k + 1) * 32);   // overlap next chunk with compute
        const int bh  = (k & 1) * (ENT * ENT_H);
        const int ic0 = k * 32;
#pragma unroll
        for (int tap = 0; tap < 9; ++tap) {
            const int dy = tap / 3, dx = tap - dy * 3;
            const int aoff = bh + (dy * 66 + m0 + ml + dx) * ENT_H + hh * 16;
            v16bf a = make_frag(&ls[aoff], &ls[aoff + 8]);
            const unsigned short* bp = WattT + ((tap * 64 + nb + ml) * 256) + ic0 + hh * 16;
            v16bf f0 = make_frag(bp, bp + 8);
            const unsigned short* bq = bp + 16 * 256;
            v16bf f1 = make_frag(bq, bq + 8);
            c0 = __builtin_amdgcn_wmma_f32_16x16x32_bf16(false, a, false, f0, (short)0, c0, false, false);
            c1 = __builtin_amdgcn_wmma_f32_16x16x32_bf16(false, a, false, f1, (short)0, c1, false, false);
        }
    }
    // epilogue: sigmoid(conv+b) * depth -> adT NHWC bf16
    const long rowO = (((long)b * 64 + y) * 64) * 64;
#pragma unroll
    for (int i = 0; i < 8; ++i) {
        const int x = m0 + i + hh * 8;
        {
            const int n = nb + ml;
            float v = c0[i] + b_att[n];
            float att = 1.f / (1.f + __expf(-v));
            float dep = depth[((long)(b * 64 + n)) * 4096 + y * 64 + x];
            adT[rowO + (long)x * 64 + n] = f2bf(att * dep);
        }
        {
            const int n = nb + 16 + ml;
            float v = c1[i] + b_att[n];
            float att = 1.f / (1.f + __expf(-v));
            float dep = depth[((long)(b * 64 + n)) * 4096 + y * 64 + x];
            adT[rowO + (long)x * 64 + n] = f2bf(att * dep);
        }
    }
}

// ---------------- conv2: async-staged bf16 WMMA (N=16 padded) + smooth-L1 + reduction ----------------
__global__ __launch_bounds__(256) void k_conv2(const unsigned short* __restrict__ adT,
                                               const float* __restrict__ b_post,
                                               const unsigned short* __restrict__ WpostT,
                                               const float* __restrict__ flow,
                                               const unsigned short* __restrict__ zerobuf,
                                               float* __restrict__ out) {
    __shared__ __align__(16) unsigned char lbuf[2 * ENT * ENT_B + 16];
    __shared__ float red[256];
    const int tid = threadIdx.x, y = blockIdx.x, b = blockIdx.y;
    const int wv = tid >> 5, lane = tid & 31;
    const int m0 = wv * 16;                 // waves 0-3 compute; all 8 waves stage
    const int ml = lane & 15, hh = lane >> 4;
    const unsigned lds_base = (unsigned)(unsigned long long)(void*)lbuf;
    const unsigned dump     = lds_base + 2u * ENT * ENT_B;
    const unsigned short* ls = (const unsigned short*)lbuf;

    unsigned sl[4];
    long     sg[4];
#pragma unroll
    for (int s = 0; s < 4; ++s) {
        int p = (wv * 4 + s) * 32 + lane;
        int e = p >> 2, q = p & 3;
        bool valid = (e < ENT);
        int r  = valid ? (e / 66) : 0;
        int cc = valid ? (e - r * 66) : 0;
        int yy = y - 1 + r, xx = cc - 1;
        bool inb = valid && ((unsigned)yy < 64u) && ((unsigned)xx < 64u);
        sl[s] = valid ? (lds_base + (unsigned)(e * ENT_B + q * 16)) : dump;
        sg[s] = inb ? ((((long)b * 64 + yy) * 64 + xx) * 64 + q * 8) : -1;
    }
    auto stage = [&](int bsel, int ic0) {
        unsigned boff = (unsigned)(bsel * ENT * ENT_B);
#pragma unroll
        for (int s = 0; s < 4; ++s) {
            const void* g = (sg[s] >= 0) ? (const void*)(adT + sg[s] + ic0)
                                         : (const void*)zerobuf;
            unsigned l = (sl[s] != dump) ? (sl[s] + boff) : dump;
            async_cp16(l, g);
        }
    };

    v8f c = {};
    stage(0, 0);
    for (int k = 0; k < 2; ++k) {
        wait_async0();
        __syncthreads();
        if (k < 1) stage(1, 32);
        if (wv < 4) {
            const int bh  = (k & 1) * (ENT * ENT_H);
            const int ic0 = k * 32;
#pragma unroll
            for (int tap = 0; tap < 9; ++tap) {
                const int dy = tap / 3, dx = tap - dy * 3;
                const int aoff = bh + (dy * 66 + m0 + ml + dx) * ENT_H + hh * 16;
                v16bf a = make_frag(&ls[aoff], &ls[aoff + 8]);
                const unsigned short* bp = WpostT + ((tap * 16 + ml) * 64) + ic0 + hh * 16;
                v16bf f = make_frag(bp, bp + 8);
                c = __builtin_amdgcn_wmma_f32_16x16x32_bf16(false, a, false, f, (short)0, c, false, false);
            }
        }
    }
    float s = 0.f;
    if (wv < 4) {
        const int n = ml;
#pragma unroll
        for (int i = 0; i < 8; ++i) {
            if (n < 10) {
                const int x = m0 + i + hh * 8;
                float v = c[i] + b_post[n];
                float t = flow[((long)(b * 10 + n)) * 4096 + y * 64 + x];
                float d  = v - t;
                float a_ = fabsf(d);
                s += (a_ < 0.01f) ? (0.5f * d * d * 100.f) : (a_ - 0.005f);
            }
        }
    }
    red[tid] = s;
    __syncthreads();
    for (int st = 128; st > 0; st >>= 1) {
        if (tid < st) red[tid] += red[tid + st];
        __syncthreads();
    }
    if (tid == 0) atomicAdd(out, red[0] * (1.f / (64.f * 64.f * 32.f * 10.f)));
}

extern "C" void kernel_launch(void* const* d_in, const int* in_sizes, int n_in,
                              void* d_out, int out_size, void* d_ws, size_t ws_size,
                              hipStream_t stream) {
    const float* feat   = (const float*)d_in[0];
    const float* depth  = (const float*)d_in[1];
    const float* gts    = (const float*)d_in[2];
    const float* W_att  = (const float*)d_in[3];
    const float* b_att  = (const float*)d_in[4];
    const float* W_post = (const float*)d_in[5];
    const float* b_post = (const float*)d_in[6];
    float* out = (float*)d_out;

    char* ws = (char*)d_ws;
    size_t off = 0;
    auto alloc = [&](size_t bytes) -> void* {
        void* p = ws + off;
        off = (off + bytes + 255) & ~(size_t)255;
        return p;
    };
    unsigned short* WattT  = (unsigned short*)alloc((size_t)9 * 64 * 256 * 2);
    unsigned short* WpostT = (unsigned short*)alloc((size_t)9 * 16 * 64 * 2);
    float* Bn   = (float*)alloc((size_t)32 * 64 * 64 * 4);
    float* Bs   = (float*)alloc((size_t)32 * 64 * 64 * 4);
    float* Bw   = (float*)alloc((size_t)32 * 64 * 64 * 4);
    float* Be   = (float*)alloc((size_t)32 * 64 * 64 * 4);
    float* flow = (float*)alloc((size_t)32 * 10 * 64 * 64 * 4);
    unsigned short* adT   = (unsigned short*)alloc((size_t)32 * 64 * 64 * 64 * 2);
    unsigned short* featT = (unsigned short*)alloc((size_t)32 * 64 * 64 * 256 * 2);
    unsigned short* zerobuf = (unsigned short*)alloc(256);

    hipMemsetAsync(d_out, 0, sizeof(float) * (out_size > 0 ? out_size : 1), stream);
    hipMemsetAsync(zerobuf, 0, 256, stream);

    dim3 grid(64, 32);
    k_nhwc      <<<grid, 256, 0, stream>>>(feat, featT);
    k_prep_watt <<<(9 * 64 * 256 + 255) / 256, 256, 0, stream>>>(W_att, WattT);
    k_prep_wpost<<<(9 * 16 * 64 + 255) / 256, 256, 0, stream>>>(W_post, WpostT);
    k_border    <<<(32 * 64 * 64) / 256, 256, 0, stream>>>(gts, Bn, Bs, Bw, Be);
    k_flow      <<<(32 * 64 * 64) / 256, 256, 0, stream>>>(Bn, Bs, Bw, Be, flow);

    k_conv1<<<grid, 256, 0, stream>>>(featT, depth, b_att, WattT, zerobuf, adT);
    k_conv2<<<grid, 256, 0, stream>>>(adT, b_post, WpostT, flow, zerobuf, out);
}
